// SDFNetwork_12292196401843
// MI455X (gfx1250) — matrix operand order repaired and checked by reading, not compiled, not run on hardware
//
#include <hip/hip_runtime.h>
#include <hip/hip_bf16.h>

typedef __attribute__((ext_vector_type(16))) _Float16 v16h;
typedef __attribute__((ext_vector_type(8)))  _Float16 v8h;
typedef __attribute__((ext_vector_type(8)))  float    v8f;

#define HG_L     16
#define HG_T     (1 << 18)          // 262144 entries per level
#define HG_MASK  (HG_T - 1)
#define P1H      2654435761u
#define P2H      805459861u

#define KPAD1    64                 // 35 -> 64 (two 32-K steps)
#define NCOL1    128
#define KPAD2    128
#define NCOL2    144                // 129 -> 144 (nine 16-col tiles)
#define W1H_ELEMS (NCOL1 * KPAD1)   // 8192 halfs
#define W2H_ELEMS (NCOL2 * KPAD2)   // 18432 halfs

// ---------------------------------------------------------------------------
// Weight pre-conversion: fp32 row-major W1[35,128], W2[128,129]  ->
// f16, B-operand-friendly layout [n][K] with K zero-padded.
// ---------------------------------------------------------------------------
__global__ void hg_convert_weights(const float* __restrict__ W1,
                                   const float* __restrict__ W2,
                                   _Float16* __restrict__ wh) {
    int idx = blockIdx.x * blockDim.x + threadIdx.x;
    if (idx < W1H_ELEMS) {
        int n = idx >> 6;           // 0..127
        int k = idx & 63;           // 0..63
        float v = (k < 35) ? W1[k * 128 + n] : 0.0f;
        wh[idx] = (_Float16)v;
    } else if (idx < W1H_ELEMS + W2H_ELEMS) {
        int j = idx - W1H_ELEMS;
        int n = j >> 7;             // 0..143
        int k = j & 127;            // 0..127
        float v = (n < 129) ? W2[k * 129 + n] : 0.0f;
        wh[W1H_ELEMS + j] = (_Float16)v;
    }
}

// ---------------------------------------------------------------------------
// Fragment builders (CDNA5 WMMA f16 VGPR layouts, wave32)
// ---------------------------------------------------------------------------
__device__ __forceinline__ v16h hg_cat8(v8h lo, v8h hi) {
    return __builtin_shufflevector(lo, hi, 0, 1, 2, 3, 4, 5, 6, 7,
                                           8, 9, 10, 11, 12, 13, 14, 15);
}

// A 16x32 f16: lane (0-15) holds row M=lane, K = {0..7, 16..23};
//              lane (16-31) holds row M=lane-16, K = {8..15, 24..31}
__device__ __forceinline__ v16h hg_fragA(const _Float16* rowp, int kbase, int lane) {
    int b = (lane >> 4) << 3;                           // 0 or 8
    v8h lo = *(const v8h*)(rowp + kbase + b);
    v8h hi = *(const v8h*)(rowp + kbase + b + 16);
    return hg_cat8(lo, hi);
}

// B 32x16 f16: lane (0-15) holds col N=lane, K = {0..7, 16..23};
//              lane (16-31) holds col N=lane-16, K = {8..15, 24..31}
__device__ __forceinline__ v16h hg_fragB(const _Float16* colp, int kbase, int lane) {
    int b = (lane >> 4) << 3;                           // 0 or 8
    v8h lo = *(const v8h*)(colp + kbase + b);
    v8h hi = *(const v8h*)(colp + kbase + b + 16);
    return hg_cat8(lo, hi);
}

__device__ __forceinline__ v8f hg_wmma(v16h a, v16h b, v8f c) {
    return __builtin_amdgcn_wmma_f32_16x16x32_f16(
        /*neg_a=*/false, a, /*neg_b=*/false, b,
        /*c_mod=*/(short)0, c, /*reuse_a=*/false, /*reuse_b=*/false);
}

// ---------------------------------------------------------------------------
// Fused HashGrid encode + 2-layer MLP.
// Block: 256 threads (8 waves), 128 points. Wave w owns points [16w,16w+16).
// ---------------------------------------------------------------------------
__launch_bounds__(256)
__global__ void hg_fused_kernel(const float* __restrict__ x,
                                const float* __restrict__ tables,
                                const _Float16* __restrict__ wh,
                                float* __restrict__ out,
                                int N) {
    // Rows [p][0..63]: f16 padded encoding (K for layer 1).
    // Overwritten in-place by h = relu(enc@W1) -> [p][0..127] (K for layer 2).
    __shared__ __align__(32) _Float16 s_eh[128 * 128];

    const int tid  = threadIdx.x;
    const int lane = tid & 31;
    const int wave = tid >> 5;

    // ---------------- encoding: thread t -> point t>>1, 8 levels -----------
    {
        const int p    = tid >> 1;
        const int lvl0 = (tid & 1) * 8;
        long row = (long)blockIdx.x * 128 + p;
        if (row >= N) row = N - 1;                 // clamp (stores guarded later)

        const float px = x[row * 3 + 0];
        const float py = x[row * 3 + 1];
        const float pz = x[row * 3 + 2];

        _Float16* er = &s_eh[p * 128];
        if ((tid & 1) == 0) {
            er[0] = (_Float16)px; er[1] = (_Float16)py; er[2] = (_Float16)pz;
        } else {
            #pragma unroll
            for (int c = 35; c < 64; ++c) er[c] = (_Float16)0.0f;
        }

        #pragma unroll
        for (int li = 0; li < 8; ++li) {
            const int l = lvl0 + li;
            // res = floor(16 * sqrt(2)^l): exact powers of two for even l
            const float res =
                floorf((float)(16 << (l >> 1)) * ((l & 1) ? 1.41421356237f : 1.0f));

            const float xs0 = px * res, xs1 = py * res, xs2 = pz * res;
            const float f0 = floorf(xs0), f1 = floorf(xs1), f2 = floorf(xs2);
            const float t0 = xs0 - f0, t1 = xs1 - f1, t2 = xs2 - f2;
            const float w0 = t0 * t0 * (3.0f - 2.0f * t0);
            const float w1 = t1 * t1 * (3.0f - 2.0f * t1);
            const float w2 = t2 * t2 * (3.0f - 2.0f * t2);
            const unsigned ux = (unsigned)(int)f0;
            const unsigned uy = (unsigned)(int)f1;
            const unsigned uz = (unsigned)(int)f2;

            const float2* tl = ((const float2*)tables) + ((size_t)l << 18);

            float a0 = 0.0f, a1 = 0.0f;
            #pragma unroll
            for (int c = 0; c < 8; ++c) {
                const int bi = (c >> 2) & 1, bj = (c >> 1) & 1, bk = c & 1;
                const unsigned h = ((ux + (unsigned)bi) ^
                                    ((uy + (unsigned)bj) * P1H) ^
                                    ((uz + (unsigned)bk) * P2H)) & HG_MASK;
                const float2 f = tl[h];
                const float wt = (bi ? w0 : 1.0f - w0) *
                                 (bj ? w1 : 1.0f - w1) *
                                 (bk ? w2 : 1.0f - w2);
                a0 = fmaf(wt, f.x, a0);
                a1 = fmaf(wt, f.y, a1);
            }
            er[3 + 2 * l]     = (_Float16)a0;
            er[3 + 2 * l + 1] = (_Float16)a1;
        }
    }
    __syncthreads();   // defensive; data flow is wave-local by construction

    // ---------------- layer 1: [16x64] @ [64x128] -> relu -> LDS -----------
    const _Float16* w1h = wh;
    const _Float16* w2h = wh + W1H_ELEMS;

    const int p0 = wave * 16;
    const int nc = lane & 15;                       // lane's N (and A-row M)
    const _Float16* arow = &s_eh[(p0 + nc) * 128];

    const v16h a0 = hg_fragA(arow, 0, lane);
    const v16h a1 = hg_fragA(arow, 32, lane);

    const int mbase = p0 + ((lane >> 4) << 3);      // C: rows M = mbase + r

    #pragma unroll
    for (int t = 0; t < 8; ++t) {
        const _Float16* bcol = &w1h[(t * 16 + nc) * KPAD1];
        v16h b0 = hg_fragB(bcol, 0, lane);
        v16h b1 = hg_fragB(bcol, 32, lane);
        v8f acc = {};
        acc = hg_wmma(a0, b0, acc);
        acc = hg_wmma(a1, b1, acc);
        #pragma unroll
        for (int r = 0; r < 8; ++r) {
            float v = acc[r] > 0.0f ? acc[r] : 0.0f;        // ReLU
            s_eh[(mbase + r) * 128 + t * 16 + nc] = (_Float16)v;
        }
    }

    // ---------------- layer 2: [16x128] @ [128x144] -> out -----------------
    v16h a2[4];
    #pragma unroll
    for (int ks = 0; ks < 4; ++ks) a2[ks] = hg_fragA(arow, ks * 32, lane);

    const long baseRow = (long)blockIdx.x * 128 + (long)mbase;

    #pragma unroll
    for (int t = 0; t < 9; ++t) {
        const _Float16* bcol = &w2h[(t * 16 + nc) * KPAD2];
        v8f acc = {};
        #pragma unroll
        for (int ks = 0; ks < 4; ++ks) {
            v16h b = hg_fragB(bcol, ks * 32, lane);
            acc = hg_wmma(a2[ks], b, acc);
        }
        const int col = t * 16 + nc;
        if (col < 129) {
            #pragma unroll
            for (int r = 0; r < 8; ++r) {
                const long grow = baseRow + r;
                if (grow < N) out[grow * 129 + col] = acc[r];
            }
        }
    }
}

// ---------------------------------------------------------------------------
// Launch
// ---------------------------------------------------------------------------
extern "C" void kernel_launch(void* const* d_in, const int* in_sizes, int n_in,
                              void* d_out, int out_size, void* d_ws, size_t ws_size,
                              hipStream_t stream) {
    const float* x      = (const float*)d_in[0];   // [N,3]
    const float* tables = (const float*)d_in[1];   // [16, 262144, 2]
    const float* W1     = (const float*)d_in[2];   // [35,128]
    const float* W2     = (const float*)d_in[3];   // [128,129]
    float* out = (float*)d_out;                    // [N,129]
    _Float16* wh = (_Float16*)d_ws;                // 8192 + 18432 halfs (~52 KB)

    const int N = in_sizes[0] / 3;

    const int convElems = W1H_ELEMS + W2H_ELEMS;
    hg_convert_weights<<<(convElems + 255) / 256, 256, 0, stream>>>(W1, W2, wh);

    const int blocks = (N + 127) / 128;
    hg_fused_kernel<<<blocks, 256, 0, stream>>>(x, tables, wh, out, N);
}
